// ObjectProposalGenerator_53652731461788
// MI455X (gfx1250) — matrix operand order repaired and checked by reading, not compiled, not run on hardware
//
#include <hip/hip_runtime.h>
#include <hip/hip_bf16.h>
#include <math.h>

// ---------------- problem constants ----------------
#define BATCH 16
#define CIN   256
#define MID   128
#define HH    128
#define WW    128
#define HW    (HH*WW)          // 16384
#define NPROP 100
#define NOUT  256              // fused out-channels: [0..127]=obj, [128..255]=box
#define KTOT  (9*CIN)          // 2304
#define PADW  130              // padded spatial extent (1-px zero halo)
#define ROWE  (PADW*CIN)       // 33280 elements per padded (h) row of one image
#define LDS_ROW_HALFS 264      // 256 payload + 8 TDM-pad halfs -> conflict-free stride
#define LDS_BYTES (3*PADW*LDS_ROW_HALFS*2)   // 205,920 B dynamic LDS

typedef __attribute__((ext_vector_type(16))) __bf16 v16bf;
typedef __attribute__((ext_vector_type(8)))  float  v8f;
typedef __attribute__((ext_vector_type(4)))  unsigned int v4u;
typedef __attribute__((ext_vector_type(8)))  int v8i;
typedef __attribute__((ext_vector_type(4)))  int v4i;

union FragU { uint4 q[2]; v16bf v; };

// ---------------- workspace layout (bytes) ----------------
#define WS_FEATPAD 0ull                       // bf16 NHWC padded: 16*130*130*256*2 = 138,444,800
#define WS_WPACK   138444800ull               // packed bf16 B-fragments: 2304*256*2
#define WS_SCORES  139624448ull               // f32 [B][HW]
#define WS_DELTAS  140673024ull               // f32 [B][4][HW]
#define WS_TOPV    144867328ull               // f32 [B][100]
#define WS_TOPI    144873728ull               // i32 [B][100]

// ===========================================================
// Kernel 0: zero the padded feature buffer (halo must be 0)
// ===========================================================
__global__ __launch_bounds__(256)
void zero_featpad(uint4* __restrict__ p) {
    size_t i = (size_t)blockIdx.x * 256 + threadIdx.x;   // grid sized exactly
    uint4 z; z.x = z.y = z.z = z.w = 0u;
    p[i] = z;
}

// ===========================================================
// Kernel 1: NCHW f32 -> padded NHWC bf16 (tiled transpose)
// grid = B*H*(CIN/32)*(WW/32) blocks of (32,8)
// ===========================================================
__global__ __launch_bounds__(256)
void feat_to_bf16_nhwc(const float* __restrict__ feat, unsigned short* __restrict__ featPad) {
    __shared__ float tile[32][33];
    int blk = blockIdx.x;
    int wb = blk & 3;  blk >>= 2;
    int cb = blk & 7;  blk >>= 3;
    int h  = blk & 127; blk >>= 7;
    int b  = blk;
    int tx = threadIdx.x, ty = threadIdx.y;
#pragma unroll
    for (int k = 0; k < 4; ++k) {
        int cil = ty + k*8;
        tile[cil][tx] = feat[(((size_t)b*CIN + cb*32 + cil)*HH + h)*WW + wb*32 + tx];
    }
    __syncthreads();
#pragma unroll
    for (int k = 0; k < 4; ++k) {
        int wl = ty + k*8;
        __hip_bfloat16 hv = __float2bfloat16(tile[tx][wl]);
        featPad[(((size_t)b*PADW + (h+1))*PADW + (wb*32 + wl + 1))*CIN + cb*32 + tx] =
            *reinterpret_cast<unsigned short*>(&hv);
    }
}

// ===========================================================
// Kernel 2: pack w1o|w1b into WMMA B-fragment order (bf16)
// layout: ((kb*16+nt)*32+lane)*16+e ; k=kb*32+8*(lane>>4)+(e&7)+16*(e>>3), K=(kh*3+kw)*256+ci, n=nt*16+(lane&15)
// ===========================================================
__global__ __launch_bounds__(256)
void pack_weights(const float* __restrict__ w1o, const float* __restrict__ w1b,
                  unsigned short* __restrict__ wpack) {
    int idx  = blockIdx.x * 256 + threadIdx.x;   // < 2304*256
    int e    = idx & 15;
    int lane = (idx >> 4) & 31;
    int f    = idx >> 9;
    int nt   = f & 15;
    int kb   = f >> 4;
    int k    = kb*32 + 8*(lane >> 4) + (e & 7) + 16*(e >> 3);
    int pos  = k >> 8;          // kh*3+kw
    int ci   = k & 255;
    int kh   = pos / 3, kw = pos - 3*kh;
    int n    = nt*16 + (lane & 15);
    float v = (n < MID) ? w1o[(((size_t)n*CIN + ci)*3 + kh)*3 + kw]
                        : w1b[(((size_t)(n-MID)*CIN + ci)*3 + kh)*3 + kw];
    __hip_bfloat16 hv = __float2bfloat16(v);
    wpack[idx] = *reinterpret_cast<unsigned short*>(&hv);
}

// ===========================================================
// Kernel 3: full-row implicit-GEMM conv (M=128,N=256,K=2304)
// TDM stages the 3x130x256 bf16 halo tile into LDS with HW
// padding (264-half rows); 8 waves split M, each owns all 16
// N-tiles (16 accumulators). B fragments flow through a 4-deep
// rotating register buffer so each WMMA consumes a fragment
// loaded 4 steps earlier (3 fragment-loads always in flight).
// Fused bias+ReLU+1x1 heads via LDS.
// grid = B*H, block = 256, dynamic LDS = LDS_BYTES
// ===========================================================
__global__ __launch_bounds__(256)
void conv_gemm_heads(const unsigned short* __restrict__ featPad,
                     const unsigned short* __restrict__ wpack,
                     const float* __restrict__ b1o, const float* __restrict__ b1b,
                     const float* __restrict__ w2o, const float* __restrict__ b2o,
                     const float* __restrict__ w2b, const float* __restrict__ b2b,
                     float* __restrict__ scores, float* __restrict__ deltas) {
    extern __shared__ char dynLDS[];
    unsigned short* ldsA  = reinterpret_cast<unsigned short*>(dynLDS);
    float*          ldsMid = reinterpret_cast<float*>(dynLDS);

    int blk = blockIdx.x;
    int h = blk & 127;
    int b = blk >> 7;
    int tid  = threadIdx.x;
    int lane = tid & 31;
    int wave = tid >> 5;
    int pl   = lane & 15;        // pixel within wave's 16-pixel M sub-tile
    int hi   = lane >> 4;        // K-half selector of the A fragment layout

    // ---- TDM: one 2D descriptor moves rows (h-1..h+1) of the padded
    //      image (each 33280 contiguous bf16) into LDS, inserting
    //      4 pad dwords every 128 dwords -> 264-half LDS row stride ----
    if (tid < 32) {   // wave 0 issues the DMA (wave-uniform branch)
        unsigned ldsOff = (unsigned)(size_t)(&ldsA[0]);
        unsigned long long ga =
            (unsigned long long)(size_t)(featPad + ((size_t)b*PADW + h) * ROWE);
        v4u g0;
        g0.x = 1u;                                   // count=1, user descriptor
        g0.y = ldsOff;                               // LDS byte address
        g0.z = (unsigned)(ga & 0xFFFFFFFFull);       // global_addr[31:0]
        g0.w = (unsigned)((ga >> 32) & 0x1FFFFFFull) // global_addr[56:32]
             | (2u << 30);                           // type=2 ("image")
        unsigned dim0 = (unsigned)ROWE;              // 33280 elements
        v8i g1;
        g1[0] = (int)((1u << 16)     // data_size: 2 bytes
                    | (1u << 20)     // pad_enable
                    | (6u << 22)     // pad_interval: 128 dwords
                    | (3u << 25));   // pad_amount: 4 dwords
        g1[1] = (int)((dim0 & 0xFFFFu) << 16);               // tensor_dim0 lo16
        g1[2] = (int)(((dim0 >> 16) & 0xFFFFu)               // tensor_dim0 hi16
                    | ((unsigned)(BATCH*PADW) & 0xFFFFu) << 16); // tensor_dim1 lo16
        g1[3] = (int)((dim0 & 0xFFFFu) << 16);               // tile_dim0 = 33280
        g1[4] = 3;                                           // tile_dim1 = 3 rows
        g1[5] = (int)dim0;                                   // tensor_dim0_stride lo32
        g1[6] = 0;
        g1[7] = 0;
        v4i gz4; gz4[0] = gz4[1] = gz4[2] = gz4[3] = 0;      // groups 2/3 unused (tile_dim2=0)
        v8i gz8;
        gz8[0] = gz8[1] = gz8[2] = gz8[3] = 0;
        gz8[4] = gz8[5] = gz8[6] = gz8[7] = 0;
        __builtin_amdgcn_tensor_load_to_lds(g0, g1, gz4, gz4, gz8, 0);
        __builtin_amdgcn_s_wait_tensorcnt((short)0);
    }
    __syncthreads();

    // ---- K loop: 72 k-blocks; per block 1 A-frag, 16 B-frags, 16 WMMAs ----
    v8f acc[16];
#pragma unroll
    for (int nt = 0; nt < 16; ++nt)
        acc[nt] = (v8f){0.f,0.f,0.f,0.f,0.f,0.f,0.f,0.f};

    const uint4* wp4 = reinterpret_cast<const uint4*>(wpack);
    for (int pos = 0; pos < 9; ++pos) {
        int rr = pos / 3, kw = pos - 3*rr;
        int arow = (rr*PADW + wave*16 + pl + kw) * LDS_ROW_HALFS;
#pragma unroll 1
        for (int cb = 0; cb < 8; ++cb) {
            int kb = pos*8 + cb;
            FragU a;
            const uint4* ap = reinterpret_cast<const uint4*>(&ldsA[arow + cb*32 + hi*8]);
            a.q[0] = ap[0];                     // K offsets +0..7   (halfs)
            a.q[1] = ap[2];                     // K offsets +16..23 (halfs)
            const uint4* bp = wp4 + ((size_t)kb*16*32 + lane)*2;  // nt=0 frag, this lane
            // 4-deep rotating B buffer: bb[1..3] stay live while bb[0]
            // is consumed+refilled, so RA must keep 4 distinct register
            // sets -> each WMMA's operand was loaded 4 iterations ago.
            FragU bb[4];
#pragma unroll
            for (int i = 0; i < 4; ++i) {
                bb[i].q[0] = bp[i*64];          // 64 uint4 per fragment (32 lanes x 32B)
                bb[i].q[1] = bp[i*64 + 1];
            }
#pragma unroll
            for (int nt = 0; nt < 16; ++nt) {
                acc[nt] = __builtin_amdgcn_wmma_f32_16x16x32_bf16(
                              false, a.v, false, bb[nt & 3].v, (short)0, acc[nt], false, false);
                if (nt < 12) {
                    bb[nt & 3].q[0] = bp[(nt+4)*64];
                    bb[nt & 3].q[1] = bp[(nt+4)*64 + 1];
                }
            }
        }
    }

    // ---- epilogue: bias + ReLU -> LDS mid tile (aliases A buffer) ----
    __syncthreads();                            // all waves done reading ldsA
    int ncol = lane & 15;
#pragma unroll
    for (int nt = 0; nt < 16; ++nt) {
        int n = nt*16 + ncol;
        float bias = (n < MID) ? b1o[n] : b1b[n - MID];
#pragma unroll
        for (int r = 0; r < 8; ++r) {
            int prow = wave*16 + r + hi*8;      // C/D layout: M = r + (lane>=16 ? 8:0)
            float v = acc[nt][r] + bias;
            ldsMid[prow*NOUT + n] = v > 0.f ? v : 0.f;
        }
    }
    __syncthreads();

    // ---- fused 1x1 heads: 128 pixels x (1 sigmoid score + 4 deltas) ----
    for (int task = tid; task < 128*5; task += 256) {
        int pp = task / 5, o = task - 5*pp;
        int gidx = h*WW + pp;
        if (o == 0) {
            float s = b2o[0];
            for (int n = 0; n < MID; ++n) s += ldsMid[pp*NOUT + n] * w2o[n];
            scores[(size_t)b*HW + gidx] = 1.0f / (1.0f + __expf(-s));
        } else {
            int j = o - 1;
            float s = b2b[j];
            for (int n = 0; n < MID; ++n) s += ldsMid[pp*NOUT + MID + n] * w2b[j*MID + n];
            deltas[((size_t)b*4 + j)*HW + gidx] = s;
        }
    }
}

// ===========================================================
// Kernel 4: top-100 per batch (iterative argmax, tie -> min idx)
// ===========================================================
__global__ __launch_bounds__(256)
void topk_kernel(float* __restrict__ scores, float* __restrict__ topv, int* __restrict__ topi) {
    int b = blockIdx.x;
    float* s = scores + (size_t)b*HW;
    __shared__ float bv[256];
    __shared__ int   bi[256];
    for (int it = 0; it < NPROP; ++it) {
        float best = -INFINITY; int bidx = HW;
        for (int j = threadIdx.x; j < HW; j += 256) {
            float v = s[j];
            if (v > best || (v == best && j < bidx)) { best = v; bidx = j; }
        }
        bv[threadIdx.x] = best; bi[threadIdx.x] = bidx;
        __syncthreads();
        if (threadIdx.x == 0) {
            float B = -INFINITY; int I = HW;
            for (int t = 0; t < 256; ++t)
                if (bv[t] > B || (bv[t] == B && bi[t] < I)) { B = bv[t]; I = bi[t]; }
            topv[b*NPROP + it] = B;
            topi[b*NPROP + it] = I;
            s[I] = -INFINITY;
        }
        __syncthreads();
    }
}

// ===========================================================
// Kernel 5: bbox decode + feature gather + output assembly
// ===========================================================
__global__ __launch_bounds__(256)
void finalize_kernel(const float* __restrict__ feat, const float* __restrict__ deltas,
                     const float* __restrict__ topv, const int* __restrict__ topi,
                     float* __restrict__ out) {
    int blk = blockIdx.x;
    int b = blk / NPROP, i = blk - b*NPROP;
    int idx = topi[b*NPROP + i];
    size_t row = (size_t)(b*NPROP + i) * 261;
    if (threadIdx.x == 0) {
        float val = topv[b*NPROP + i];
        int y = idx >> 7, x = idx & 127;
        float dx = deltas[((size_t)b*4 + 0)*HW + idx];
        float dy = deltas[((size_t)b*4 + 1)*HW + idx];
        float dw = deltas[((size_t)b*4 + 2)*HW + idx];
        float dh = deltas[((size_t)b*4 + 3)*HW + idx];
        float cx = (float)x*8.0f + dx*8.0f;
        float cy = (float)y*8.0f + dy*8.0f;
        float bw = __expf(dw)*8.0f;
        float bh = __expf(dh)*8.0f;
        out[row+0] = cx - bw*0.5f;
        out[row+1] = cy - bh*0.5f;
        out[row+2] = cx + bw*0.5f;
        out[row+3] = cy + bh*0.5f;
        out[row+4] = val;
        float* loc = out + (size_t)BATCH*NPROP*261;
        loc[(b*NPROP + i)*2 + 0] = (float)x;
        loc[(b*NPROP + i)*2 + 1] = (float)y;
    }
    int c = threadIdx.x;
    out[row + 5 + c] = feat[((size_t)b*CIN + c)*HW + idx];
}

// ===========================================================
extern "C" void kernel_launch(void* const* d_in, const int* in_sizes, int n_in,
                              void* d_out, int out_size, void* d_ws, size_t ws_size,
                              hipStream_t stream) {
    const float* features = (const float*)d_in[0];
    const float* w1o      = (const float*)d_in[1];
    const float* b1o      = (const float*)d_in[2];
    const float* w2o      = (const float*)d_in[3];
    const float* b2o      = (const float*)d_in[4];
    const float* w1b      = (const float*)d_in[5];
    const float* b1b      = (const float*)d_in[6];
    const float* w2b      = (const float*)d_in[7];
    const float* b2b      = (const float*)d_in[8];

    char* ws = (char*)d_ws;
    unsigned short* featPad = (unsigned short*)(ws + WS_FEATPAD);
    unsigned short* wpack   = (unsigned short*)(ws + WS_WPACK);
    float* scores = (float*)(ws + WS_SCORES);
    float* deltas = (float*)(ws + WS_DELTAS);
    float* topv   = (float*)(ws + WS_TOPV);
    int*   topi   = (int*)  (ws + WS_TOPI);
    float* out    = (float*)d_out;

    // padded feature buffer: 138,444,800 B = 8,652,800 uint4
    zero_featpad<<<33800, 256, 0, stream>>>((uint4*)featPad);
    feat_to_bf16_nhwc<<<BATCH*HH*8*4, dim3(32,8), 0, stream>>>(features, featPad);
    pack_weights<<<(KTOT*NOUT)/256, 256, 0, stream>>>(w1o, w1b, wpack);
    conv_gemm_heads<<<BATCH*HH, 256, LDS_BYTES, stream>>>(
        featPad, wpack, b1o, b1b, w2o, b2o, w2b, b2b, scores, deltas);
    topk_kernel<<<BATCH, 256, 0, stream>>>(scores, topv, topi);
    finalize_kernel<<<BATCH*NPROP, 256, 0, stream>>>(features, deltas, topv, topi, out);
}
